// ProteinGAT_28656021799363
// MI455X (gfx1250) — compile-verified, matrix operand order checked by reading, not run on hardware
//
#include <hip/hip_runtime.h>

#define Nn 4096
#define Dd 128
#define Hh 8
#define QK_SCALE 0.17677669529663687f  // 1/sqrt(32)

typedef __bf16 bf16x16 __attribute__((ext_vector_type(16)));
typedef float  f32x8   __attribute__((ext_vector_type(8)));

// CDNA5 16-bit A/B operand layout for V_WMMA_*_16X16X32_*:
// lane L holds row/col (L%16); kb=(L>>4)*8; slot i<8 -> k=kb+i, i>=8 -> k=16+kb+(i-8)
__device__ __forceinline__ bf16x16 load_tile_bf16(const __bf16* __restrict__ row, int kb) {
  bf16x16 a;
#pragma unroll
  for (int i = 0; i < 8; ++i) a[i] = row[kb + i];
#pragma unroll
  for (int i = 0; i < 8; ++i) a[8 + i] = row[16 + kb + i];
  return a;
}

__device__ __forceinline__ f32x8 wmma_bf16(bf16x16 a, bf16x16 b, f32x8 c) {
  return __builtin_amdgcn_wmma_f32_16x16x32_bf16(false, a, false, b, (short)0, c, false, false);
}

__device__ __forceinline__ float leaky(float x) { return x > 0.f ? x : 0.1f * x; }

// ---------------- f = leaky(fres @ W_emb) ----------------
__global__ void k_embed(const float* __restrict__ fres, const float* __restrict__ Wemb,
                        float* __restrict__ f0) {
  int idx = blockIdx.x * blockDim.x + threadIdx.x;  // N*128
  int n = idx >> 7, d = idx & 127;
  const float* fr = fres + n * 20;
  float acc = 0.f;
#pragma unroll
  for (int j = 0; j < 20; ++j) acc += fr[j] * Wemb[j * Dd + d];
  f0[idx] = leaky(acc);
}

// ---------------- fT_bf16[d][n] = (bf16) f0[n][d] ----------------
__global__ void k_cvtT(const float* __restrict__ f0, __bf16* __restrict__ fT) {
  int idx = blockIdx.x * blockDim.x + threadIdx.x;  // 128*N; idx = d*N + n
  int d = idx >> 12;
  int n = idx & (Nn - 1);
  fT[idx] = (__bf16)f0[n * Dd + d];
}

// ---------------- finit = f0 + DW @ f0  (bf16 WMMA, f32 acc, 2-wide col reuse) ----------------
__global__ void __launch_bounds__(256) k_dwgemm(const float* __restrict__ dw,
                                                const __bf16* __restrict__ fT,
                                                const float* __restrict__ fin,
                                                float* __restrict__ fout) {
  int lane = threadIdx.x & 31;
  int wid  = (blockIdx.x * blockDim.x + threadIdx.x) >> 5;  // 0..1023
  int iblk = wid >> 2, dpair = wid & 3;
  int rowbase = iblk * 16;
  int dcol0 = dpair * 32, dcol1 = dcol0 + 16;
  int n16 = lane & 15, half = lane >> 4, kb = half * 8;

  f32x8 acc0 = {}, acc1 = {};
  const float*  arow  = dw + (size_t)(rowbase + n16) * Nn;
  const __bf16* brow0 = fT + (size_t)(dcol0 + n16) * Nn;
  const __bf16* brow1 = fT + (size_t)(dcol1 + n16) * Nn;
  for (int jb = 0; jb < Nn; jb += 32) {
    bf16x16 a;
#pragma unroll
    for (int i = 0; i < 8; ++i) a[i] = (__bf16)arow[jb + kb + i];
#pragma unroll
    for (int i = 0; i < 8; ++i) a[8 + i] = (__bf16)arow[jb + 16 + kb + i];
    bf16x16 b0 = load_tile_bf16(brow0 + jb, kb);
    bf16x16 b1 = load_tile_bf16(brow1 + jb, kb);
    acc0 = wmma_bf16(a, b0, acc0);
    acc1 = wmma_bf16(a, b1, acc1);
  }
#pragma unroll
  for (int p = 0; p < 8; ++p) {
    int r = rowbase + p + 8 * half;
    fout[r * Dd + dcol0 + n16] = acc0[p] + fin[r * Dd + dcol0 + n16];
    fout[r * Dd + dcol1 + n16] = acc1[p] + fin[r * Dd + dcol1 + n16];
  }
}

// ---------------- q/k/v projections -> bf16 (q pre-scaled, v transposed) ----------------
__global__ void k_qkv(const float* __restrict__ f, const float* __restrict__ Wq,
                      const float* __restrict__ Wk, const float* __restrict__ Wv,
                      __bf16* __restrict__ qb, __bf16* __restrict__ kbf,
                      __bf16* __restrict__ vT) {
  int idx = blockIdx.x * blockDim.x + threadIdx.x;  // N*768
  int n = idx / 768;
  int c = idx - n * 768;
  int which = c >> 8;
  int cc = c & 255;
  const float* W = (which == 0) ? Wq : (which == 1) ? Wk : Wv;
  const float* fr = f + n * Dd;
  float acc = 0.f;
#pragma unroll 8
  for (int d = 0; d < Dd; ++d) acc += fr[d] * W[d * 256 + cc];
  int h = cc >> 5, kk = cc & 31;
  if (which == 0)      qb[((size_t)h * Nn + n) * 32 + kk] = (__bf16)(acc * QK_SCALE);
  else if (which == 1) kbf[((size_t)h * Nn + n) * 32 + kk] = (__bf16)acc;
  else                 vT[((size_t)h * 32 + kk) * Nn + n] = (__bf16)acc;
}

// ---------------- flash attention: one wave = one (head, 16-row) tile ----------------
// Software-pipelined: K/V tiles for step jb+32 are loaded before the current
// step's softmax, so global-load latency overlaps VALU work.
__global__ void __launch_bounds__(256) k_attn(const __bf16* __restrict__ qb,
                                              const __bf16* __restrict__ kbf,
                                              const __bf16* __restrict__ vT,
                                              const float* __restrict__ mask,
                                              float* __restrict__ oH) {
  __shared__ __bf16 s_p[8 * 16 * 32];  // 1KB per wave: p-tile C-layout -> A-layout staging
  int lane = threadIdx.x & 31;
  int wslot = threadIdx.x >> 5;
  int wid = blockIdx.x * 8 + wslot;     // 0..2047
  int h = wid >> 8;
  int iblk = wid & 255;
  int rowbase = iblk * 16;
  int n16 = lane & 15, half = lane >> 4, kb = half * 8;

  const __bf16* kbase  = kbf + (size_t)h * Nn * 32;
  const __bf16* vbase0 = vT + ((size_t)h * 32 + n16) * Nn;
  const __bf16* vbase1 = vT + ((size_t)h * 32 + 16 + n16) * Nn;

  bf16x16 qa = load_tile_bf16(qb + ((size_t)h * Nn + rowbase + n16) * 32, kb);

  float m[8], l[8];
  f32x8 o0 = {}, o1 = {};
#pragma unroll
  for (int p = 0; p < 8; ++p) { m[p] = -3.0e38f; l[p] = 0.f; }

  __bf16* tile = s_p + wslot * (16 * 32);
  const float* mrow = mask + (size_t)(rowbase + 8 * half) * Nn + n16;

  // preload tiles for jb = 0
  bf16x16 b0  = load_tile_bf16(kbase + (size_t)(n16) * 32, kb);
  bf16x16 b1  = load_tile_bf16(kbase + (size_t)(16 + n16) * 32, kb);
  bf16x16 vb0 = load_tile_bf16(vbase0, kb);
  bf16x16 vb1 = load_tile_bf16(vbase1, kb);

  for (int jb = 0; jb < Nn; jb += 32) {
    // prefetch tiles for the next step (wraps to 0 on the last step; harmless)
    int jn = (jb + 32) & (Nn - 1);
    bf16x16 nb0  = load_tile_bf16(kbase + (size_t)(jn + n16) * 32, kb);
    bf16x16 nb1  = load_tile_bf16(kbase + (size_t)(jn + 16 + n16) * 32, kb);
    bf16x16 nvb0 = load_tile_bf16(vbase0 + jn, kb);
    bf16x16 nvb1 = load_tile_bf16(vbase1 + jn, kb);

    f32x8 z = {};
    f32x8 s0 = wmma_bf16(qa, b0, z);
    f32x8 s1 = wmma_bf16(qa, b1, z);

#pragma unroll
    for (int p = 0; p < 8; ++p) {
      float s0p = s0[p] + mrow[(size_t)p * Nn + jb];
      float s1p = s1[p] + mrow[(size_t)p * Nn + jb + 16];
      // row max across the 16 lanes of this half-group (C layout holds a row per VGPR)
      float mx = fmaxf(s0p, s1p);
#pragma unroll
      for (int t = 1; t < 16; t <<= 1) mx = fmaxf(mx, __shfl_xor(mx, t, 32));
      float nm = fmaxf(m[p], mx);
      float al = __expf(m[p] - nm);
      m[p] = nm;
      s0p = __expf(s0p - nm);
      s1p = __expf(s1p - nm);
      float rs = s0p + s1p;
#pragma unroll
      for (int t = 1; t < 16; t <<= 1) rs += __shfl_xor(rs, t, 32);
      l[p] = l[p] * al + rs;
      o0[p] *= al;
      o1[p] *= al;
      int mr = p + 8 * half;
      tile[mr * 32 + n16] = (__bf16)s0p;
      tile[mr * 32 + 16 + n16] = (__bf16)s1p;
    }
    // per-wave LDS ordering: DS ops complete in order within a wave
    asm volatile("s_wait_dscnt 0x0" ::: "memory");
    bf16x16 pa = load_tile_bf16(tile + n16 * 32, kb);  // p-tile as 16x32 A operand
    o0 = wmma_bf16(pa, vb0, o0);
    o1 = wmma_bf16(pa, vb1, o1);

    b0 = nb0; b1 = nb1; vb0 = nvb0; vb1 = nvb1;
  }
#pragma unroll
  for (int p = 0; p < 8; ++p) {
    float inv = 1.f / l[p];
    int r = rowbase + p + 8 * half;
    oH[((size_t)h * Nn + r) * 32 + n16]      = o0[p] * inv;
    oH[((size_t)h * Nn + r) * 32 + 16 + n16] = o1[p] * inv;
  }
}

// ---------------- sum over heads ----------------
__global__ void k_headsum(const float* __restrict__ oH, float* __restrict__ osum) {
  int idx = blockIdx.x * blockDim.x + threadIdx.x;  // N*32
  float s = 0.f;
#pragma unroll
  for (int h = 0; h < Hh; ++h) s += oH[(size_t)h * Nn * 32 + idx];
  osum[idx] = s;
}

// ---------------- f = leaky(osum @ W_out + b_out) ----------------
__global__ void k_outproj(const float* __restrict__ osum, const float* __restrict__ Wout,
                          const float* __restrict__ bout, float* __restrict__ fout) {
  int idx = blockIdx.x * blockDim.x + threadIdx.x;  // N*128
  int n = idx >> 7, d = idx & 127;
  const float* orow = osum + n * 32;
  float acc = bout[d];
#pragma unroll
  for (int c = 0; c < 32; ++c) acc += orow[c] * Wout[c * Dd + d];
  fout[idx] = leaky(acc);
}

// ---------------- out = leaky([f, initial] @ W_last + b_last) ----------------
__global__ void k_final(const float* __restrict__ f, const float* __restrict__ initial,
                        const float* __restrict__ Wlast, const float* __restrict__ blast,
                        float* __restrict__ out) {
  int idx = blockIdx.x * blockDim.x + threadIdx.x;  // N*128
  int n = idx >> 7, d = idx & 127;
  const float* fr = f + n * Dd;
  const float* ir = initial + n * Dd;
  float acc = blast[d];
#pragma unroll 4
  for (int j = 0; j < Dd; ++j) acc += fr[j] * Wlast[j * Dd + d];
#pragma unroll 4
  for (int j = 0; j < Dd; ++j) acc += ir[j] * Wlast[(Dd + j) * Dd + d];
  out[idx] = leaky(acc);
}

extern "C" void kernel_launch(void* const* d_in, const int* in_sizes, int n_in,
                              void* d_out, int out_size, void* d_ws, size_t ws_size,
                              hipStream_t stream) {
  (void)in_sizes; (void)n_in; (void)out_size; (void)ws_size;
  const float* fres  = (const float*)d_in[0];
  const float* dw    = (const float*)d_in[1];
  const float* mask  = (const float*)d_in[2];
  const float* Wemb  = (const float*)d_in[3];
  const float* Wq    = (const float*)d_in[4];
  const float* Wk    = (const float*)d_in[5];
  const float* Wv    = (const float*)d_in[6];
  const float* Wout  = (const float*)d_in[7];
  const float* bout  = (const float*)d_in[8];
  const float* Wlast = (const float*)d_in[9];
  const float* blast = (const float*)d_in[10];
  float* out = (float*)d_out;

  char* ws = (char*)d_ws;
  auto alloc = [&](size_t bytes) {
    char* p = ws;
    ws += (bytes + 255) & ~(size_t)255;
    return p;
  };
  float*  f0    = (float*) alloc((size_t)Nn * Dd * 4);
  float*  finit = (float*) alloc((size_t)Nn * Dd * 4);
  float*  fit   = (float*) alloc((size_t)Nn * Dd * 4);
  __bf16* fT    = (__bf16*)alloc((size_t)Dd * Nn * 2);
  __bf16* qb    = (__bf16*)alloc((size_t)Hh * Nn * 32 * 2);
  __bf16* kbf   = (__bf16*)alloc((size_t)Hh * Nn * 32 * 2);
  __bf16* vT    = (__bf16*)alloc((size_t)Hh * 32 * Nn * 2);
  float*  oH    = (float*) alloc((size_t)Hh * Nn * 32 * 4);
  float*  osum  = (float*) alloc((size_t)Nn * 32 * 4);

  dim3 b256(256);
  k_embed<<<(Nn * Dd) / 256, b256, 0, stream>>>(fres, Wemb, f0);
  k_cvtT<<<(Nn * Dd) / 256, b256, 0, stream>>>(f0, fT);
  k_dwgemm<<<128, b256, 0, stream>>>(dw, fT, f0, finit);
  for (int it = 0; it < 3; ++it) {
    const float* fin = (it == 0) ? finit : fit;
    k_qkv<<<(Nn * 768) / 256, b256, 0, stream>>>(fin, Wq, Wk, Wv, qb, kbf, vT);
    k_attn<<<256, b256, 0, stream>>>(qb, kbf, vT, mask, oH);
    k_headsum<<<(Nn * 32) / 256, b256, 0, stream>>>(oH, osum);
    k_outproj<<<(Nn * Dd) / 256, b256, 0, stream>>>(osum, Wout, bout, fit);
  }
  k_final<<<(Nn * Dd) / 256, b256, 0, stream>>>(fit, finit, Wlast, blast, out);
}
